// DecoderBlock_85822036508828
// MI455X (gfx1250) — matrix operand order repaired and checked by reading, not compiled, not run on hardware
//
#include <hip/hip_runtime.h>

// ---------------------------------------------------------------------------
// Decoder block: S=4096, D=768, H=12, DH=64, fp32 I/O, bf16 WMMA internals.
// ---------------------------------------------------------------------------
#define S_LEN  4096
#define DMODEL 768
#define NHEADS 12
#define DHEAD  64
#define TD     2304   // 3*D
#define DFF    3072   // 4*D

typedef __attribute__((ext_vector_type(16))) __bf16 v16bf;
typedef __attribute__((ext_vector_type(8)))  __bf16 v8bf;
typedef __attribute__((ext_vector_type(8)))  float  v8f;
typedef __attribute__((ext_vector_type(4)))  unsigned int u32x4;
typedef __attribute__((ext_vector_type(8)))  int          i32x8;
typedef __attribute__((ext_vector_type(4)))  int          i32x4;

__device__ __forceinline__ v8f v8f_zero() {
    v8f z = {0.f, 0.f, 0.f, 0.f, 0.f, 0.f, 0.f, 0.f};
    return z;
}

__device__ __forceinline__ float gelu_tanh(float x) {
    // jax.nn.gelu default (approximate=True)
    float x3 = x * x * x;
    return 0.5f * x * (1.0f + tanhf(0.7978845608028654f * (x + 0.044715f * x3)));
}

// ---------------------------------------------------------------------------
// TDM: DMA a 64(N) x 32(K) bf16 tile of W into LDS (row-major, 64B rows).
// Descriptor per CDNA5 ISA ch.8 (D# group0/group1); 2D tensor so groups 2/3
// are zero/disabled. data_size=1 (2 bytes), tile_dim0=32 (K), tile_dim1=64
// (N rows), tensor_dim0_stride = K. This toolchain's builtin is the 6-arg
// form (g0, g1, g2, g3, g_extra, cpol).
// ---------------------------------------------------------------------------
__device__ __forceinline__ void tdm_load_tile_64x32(const __bf16* gsrc,
                                                    unsigned lds_off,
                                                    int K, int N) {
    const unsigned long long ga = (unsigned long long)(const void*)gsrc;
    u32x4 g0;
    g0[0] = 1u;                                            // count=1, user mode
    g0[1] = lds_off;                                       // lds_addr
    g0[2] = (unsigned)(ga & 0xFFFFFFFFull);                // global_addr[31:0]
    g0[3] = (unsigned)((ga >> 32) & 0x01FFFFFFull)         // global_addr[56:32]
          | (2u << 30);                                    // type = 2 ("image")
    i32x8 g1;
    g1[0] = (int)(1u << 16);                               // data_size = 2 bytes
    g1[1] = (int)(((unsigned)K & 0xFFFFu) << 16);          // tensor_dim0[15:0]
    g1[2] = (int)((((unsigned)K >> 16) & 0xFFFFu)          // tensor_dim0[31:16]
          | (((unsigned)N & 0xFFFFu) << 16));              // tensor_dim1[15:0]
    g1[3] = (int)(((((unsigned)N) >> 16) & 0xFFFFu)        // tensor_dim1[31:16]
          | (32u << 16));                                  // tile_dim0 = 32
    g1[4] = 64;                                            // tile_dim1=64, tile_dim2=0
    g1[5] = K;                                             // dim0_stride[31:0]
    g1[6] = 0;                                             // dim0_stride[47:32], dim1_stride lo
    g1[7] = 0;
    i32x4 gz4 = {0, 0, 0, 0};
    i32x8 gz8 = {0, 0, 0, 0, 0, 0, 0, 0};
    __builtin_amdgcn_tensor_load_to_lds(g0, g1, gz4, gz4, gz8, 0);
}

// ---------------------------------------------------------------------------
// fp32 -> bf16 conversion (weights)
// ---------------------------------------------------------------------------
__global__ void cvt_f32_bf16(const float* __restrict__ in, __bf16* __restrict__ out, int n) {
    int i = blockIdx.x * blockDim.x + threadIdx.x;
    int stride = gridDim.x * blockDim.x;
    for (; i < n; i += stride) out[i] = (__bf16)in[i];
}

// ---------------------------------------------------------------------------
// Transpose the V part of qkv (bf16 [S, 3D], V at col offset 2D) into
// vT [D, S] so attention's P.V B-fragments become contiguous 32B loads.
// Grid (S/64, D/64), block 256, 64x64 LDS tile (padded vs bank conflicts).
// ---------------------------------------------------------------------------
__global__ void __launch_bounds__(256)
transpose_v(const __bf16* __restrict__ qkv, __bf16* __restrict__ vT) {
    __shared__ __bf16 tile[64][72];
    const int tid = threadIdx.x;
    const int s0 = blockIdx.x * 64;
    const int d0 = blockIdx.y * 64;
    for (int i = tid; i < 64 * 64; i += 256) {
        const int r = i >> 6, c = i & 63;             // r: key row, c: dh col
        tile[r][c] = qkv[(size_t)(s0 + r) * TD + 2 * DMODEL + d0 + c];
    }
    __syncthreads();
    for (int i = tid; i < 64 * 64; i += 256) {
        const int r = i >> 6, c = i & 63;             // r: dh row, c: key col
        vT[(size_t)(d0 + r) * S_LEN + s0 + c] = tile[c][r];
    }
}

// ---------------------------------------------------------------------------
// LayerNorm over D=768 per row; optional fused residual add (x + addend).
// Writes fp32 (residual path) and bf16 (WMMA A operand).
// ---------------------------------------------------------------------------
__global__ void __launch_bounds__(256)
ln_fused(const float* __restrict__ x, const float* __restrict__ addend,
         const float* __restrict__ w, const float* __restrict__ b,
         float* __restrict__ outF, __bf16* __restrict__ outB) {
    const int row = blockIdx.x;
    const int tid = threadIdx.x;
    const size_t base = (size_t)row * DMODEL;

    float v[3];
    float s = 0.f, s2 = 0.f;
#pragma unroll
    for (int i = 0; i < 3; ++i) {
        const int c = tid + i * 256;
        float t = x[base + c];
        if (addend) t += addend[base + c];
        v[i] = t;
        s += t;
        s2 += t * t;
    }
#pragma unroll
    for (int m = 1; m < 32; m <<= 1) {
        s  += __shfl_xor(s,  m, 32);
        s2 += __shfl_xor(s2, m, 32);
    }
    __shared__ float red[16];
    const int wave = tid >> 5, lane = tid & 31;
    if (lane == 0) { red[wave] = s; red[8 + wave] = s2; }
    __syncthreads();
    s = 0.f; s2 = 0.f;
#pragma unroll
    for (int wv = 0; wv < 8; ++wv) { s += red[wv]; s2 += red[8 + wv]; }

    const float mu  = s * (1.0f / DMODEL);
    const float var = s2 * (1.0f / DMODEL) - mu * mu;
    const float inv = rsqrtf(var + 1e-5f);
#pragma unroll
    for (int i = 0; i < 3; ++i) {
        const int c = tid + i * 256;
        const float y = (v[i] - mu) * inv * w[c] + b[c];
        outF[base + c] = y;
        outB[base + c] = (__bf16)y;
    }
}

// ---------------------------------------------------------------------------
// C[M,N] = act(A[M,K] @ W[N,K]^T + bias) (+ residual) — bf16 WMMA, f32 acc.
// Grid: (M/128, N/64); block 256 = 8 waves; wave -> 16 rows x 64 cols.
// The B tile (64x32, shared by all 8 waves) is staged into LDS by the
// Tensor Data Mover (double-buffered, s_wait_tensorcnt + barrier);
// A streams directly from global (per-wave rows, pipelines with loadcnt).
// ---------------------------------------------------------------------------
__global__ void __launch_bounds__(256)
gemm_bf16_wmma(const __bf16* __restrict__ A, const __bf16* __restrict__ W,
               const float* __restrict__ bias, const float* __restrict__ resid,
               float* __restrict__ outF, __bf16* __restrict__ outB,
               int M, int N, int K, int act) {
    const int lane = threadIdx.x & 31;
    const int wave = threadIdx.x >> 5;
    const int half = lane >> 4;   // lane-half selects K sub-range
    const int l16  = lane & 15;
    const int rowBase = blockIdx.x * 128 + wave * 16;
    const int colBase = blockIdx.y * 64;

    __shared__ __align__(128) __bf16 btile[2][64 * 32];

    v8f acc[4];
#pragma unroll
    for (int t = 0; t < 4; ++t) acc[t] = v8f_zero();

    const int nchunks = K >> 5;
    const __bf16* Arow = A + (size_t)(rowBase + l16) * K;
    const __bf16* Wtile = W + (size_t)colBase * K;

    if (wave == 0) {
        const unsigned lds0 = (unsigned)(unsigned long long)(void*)&btile[0][0];
        tdm_load_tile_64x32(Wtile, lds0, K, N);
    }

    for (int i = 0; i < nchunks; ++i) {
        if (wave == 0) {
            if (i + 1 < nchunks) {
                const unsigned ldsn =
                    (unsigned)(unsigned long long)(void*)&btile[(i + 1) & 1][0];
                tdm_load_tile_64x32(Wtile + (size_t)(i + 1) * 32, ldsn, K, N);
                __builtin_amdgcn_s_wait_tensorcnt(1);   // chunk i landed
            } else {
                __builtin_amdgcn_s_wait_tensorcnt(0);
            }
        }
        __syncthreads();

        const int kc = i << 5;
        if (kc + 64 < K) __builtin_prefetch(Arow + kc + 64, 0, 1);

        // A-frag (16x32, contiguous k order, same as B-frag): lane row M=l16;
        // elem e -> K=kc+half*16+e
        v8bf lo = *(const v8bf*)(Arow + kc + half * 16);
        v8bf hi = *(const v8bf*)(Arow + kc + half * 16 + 8);
        v16bf afrag;
#pragma unroll
        for (int ii = 0; ii < 8; ++ii) { afrag[ii] = lo[ii]; afrag[8 + ii] = hi[ii]; }

        const __bf16* bt = &btile[i & 1][0];
#pragma unroll
        for (int t = 0; t < 4; ++t) {
            // B-frag: lane col N=l16; elem e -> K=kc+half*16+e (32B LDS read)
            v16bf bfrag = *(const v16bf*)(bt + (t * 16 + l16) * 32 + half * 16);
            acc[t] = __builtin_amdgcn_wmma_f32_16x16x32_bf16(
                false, afrag, false, bfrag, (short)0, acc[t], false, false);
        }
        __syncthreads();
    }

    // Epilogue: C layout — lane col N=l16, vgpr r -> row r + 8*half.
#pragma unroll
    for (int t = 0; t < 4; ++t) {
        const int col = colBase + t * 16 + l16;
        const float bi = bias ? bias[col] : 0.f;
#pragma unroll
        for (int r = 0; r < 8; ++r) {
            const int row = rowBase + r + 8 * half;
            float vv = acc[t][r] + bi;
            if (act == 1) vv = gelu_tanh(vv);
            const size_t idx = (size_t)row * N + col;
            if (resid) vv += resid[idx];
            if (outF) outF[idx] = vv;
            if (outB) outB[idx] = (__bf16)vv;
        }
    }
}

// ---------------------------------------------------------------------------
// Flash attention (no mask, matches reference softmax over full S).
// Block = 128 threads (4 waves); each wave owns (16 query rows, head).
// Scores and P.V via WMMA; online softmax; P transposed C->A layout via LDS.
// V comes from the pre-transposed vT [D, S] so B-frags are contiguous.
// ---------------------------------------------------------------------------
__global__ void __launch_bounds__(128)
attn_flash_wmma(const __bf16* __restrict__ qkv, const __bf16* __restrict__ vT,
                float* __restrict__ ctx) {
    const int lane = threadIdx.x & 31;
    const int wave = threadIdx.x >> 5;
    const int half = lane >> 4;
    const int l16  = lane & 15;
    const int m0 = (blockIdx.x * 4 + wave) * 16;
    const int hh = blockIdx.y;
    const float scale = 0.125f;   // 1/sqrt(DH=64)

    __shared__ __bf16 pbuf[4][16 * 32];

    // Q as two A-frags (16x32 each) covering dh = 0..63
    const __bf16* qb = qkv + (size_t)(m0 + l16) * TD + hh * DHEAD;
    v16bf aq[2];
#pragma unroll
    for (int kk = 0; kk < 2; ++kk) {
        v8bf lo = *(const v8bf*)(qb + kk * 32 + half * 16);
        v8bf hi = *(const v8bf*)(qb + kk * 32 + half * 16 + 8);
#pragma unroll
        for (int i = 0; i < 8; ++i) { aq[kk][i] = lo[i]; aq[kk][8 + i] = hi[i]; }
    }

    float mrow[8], lrow[8];
#pragma unroll
    for (int r = 0; r < 8; ++r) { mrow[r] = -3.0e38f; lrow[r] = 0.f; }
    v8f acc[4];
#pragma unroll
    for (int t = 0; t < 4; ++t) acc[t] = v8f_zero();

    for (int kc = 0; kc < S_LEN; kc += 32) {
        // --- scores: two 16x16 tiles (keys kc..kc+15, kc+16..kc+31) ---
        v8f sc[2];
#pragma unroll
        for (int j = 0; j < 2; ++j) {
            const __bf16* kb = qkv + (size_t)(kc + j * 16 + l16) * TD + DMODEL + hh * DHEAD;
            v16bf bk0 = *(const v16bf*)(kb + half * 16);        // dh 0..31 slice
            v16bf bk1 = *(const v16bf*)(kb + 32 + half * 16);   // dh 32..63 slice
            v8f z = v8f_zero();
            z = __builtin_amdgcn_wmma_f32_16x16x32_bf16(false, aq[0], false, bk0, (short)0, z, false, false);
            z = __builtin_amdgcn_wmma_f32_16x16x32_bf16(false, aq[1], false, bk1, (short)0, z, false, false);
            sc[j] = z;
        }
        // --- online softmax: row (r + 8*half) spans one 16-lane group ---
        float tm[8];
#pragma unroll
        for (int r = 0; r < 8; ++r)
            tm[r] = fmaxf(sc[0][r] * scale, sc[1][r] * scale);
#pragma unroll
        for (int m = 1; m < 16; m <<= 1) {
#pragma unroll
            for (int r = 0; r < 8; ++r)
                tm[r] = fmaxf(tm[r], __shfl_xor(tm[r], m, 16));
        }
        float alpha[8], rs[8];
#pragma unroll
        for (int r = 0; r < 8; ++r) {
            const float mn = fmaxf(mrow[r], tm[r]);
            alpha[r] = __expf(mrow[r] - mn);
            mrow[r] = mn;
            const float p0 = __expf(sc[0][r] * scale - mn);
            const float p1 = __expf(sc[1][r] * scale - mn);
            sc[0][r] = p0; sc[1][r] = p1;
            rs[r] = p0 + p1;
        }
#pragma unroll
        for (int m = 1; m < 16; m <<= 1) {
#pragma unroll
            for (int r = 0; r < 8; ++r)
                rs[r] += __shfl_xor(rs[r], m, 16);
        }
#pragma unroll
        for (int r = 0; r < 8; ++r) lrow[r] = lrow[r] * alpha[r] + rs[r];
#pragma unroll
        for (int t = 0; t < 4; ++t) {
#pragma unroll
            for (int r = 0; r < 8; ++r) acc[t][r] *= alpha[r];
        }
        // --- P: C layout -> LDS [row][key] -> reload in A layout ---
#pragma unroll
        for (int j = 0; j < 2; ++j) {
#pragma unroll
            for (int r = 0; r < 8; ++r)
                pbuf[wave][(r + 8 * half) * 32 + j * 16 + l16] = (__bf16)sc[j][r];
        }
        __syncthreads();
        v16bf ap;
        {
            v8bf lo = *(const v8bf*)(&pbuf[wave][l16 * 32 + half * 16]);
            v8bf hi = *(const v8bf*)(&pbuf[wave][l16 * 32 + half * 16 + 8]);
#pragma unroll
            for (int i = 0; i < 8; ++i) { ap[i] = lo[i]; ap[8 + i] = hi[i]; }
        }
        __syncthreads();
        // --- acc += P @ V via vT: lane col dh=t*16+l16, elem e -> key kc+half*16+e ---
#pragma unroll
        for (int t = 0; t < 4; ++t) {
            const __bf16* vb = vT + (size_t)(hh * DHEAD + t * 16 + l16) * S_LEN + kc + half * 16;
            v16bf bv = *(const v16bf*)vb;
            acc[t] = __builtin_amdgcn_wmma_f32_16x16x32_bf16(
                false, ap, false, bv, (short)0, acc[t], false, false);
        }
    }

#pragma unroll
    for (int t = 0; t < 4; ++t) {
#pragma unroll
        for (int r = 0; r < 8; ++r) {
            const int row = m0 + r + 8 * half;
            const int col = hh * DHEAD + t * 16 + l16;
            ctx[(size_t)row * DMODEL + col] = acc[t][r] / lrow[r];
        }
    }
}

// ---------------------------------------------------------------------------
// Host-side orchestration
// ---------------------------------------------------------------------------
extern "C" void kernel_launch(void* const* d_in, const int* in_sizes, int n_in,
                              void* d_out, int out_size, void* d_ws, size_t ws_size,
                              hipStream_t stream) {
    (void)in_sizes; (void)n_in; (void)out_size; (void)ws_size;

    const float* x      = (const float*)d_in[0];
    const float* ln1w   = (const float*)d_in[1];
    const float* ln1b   = (const float*)d_in[2];
    const float* w_attn = (const float*)d_in[3];
    const float* b_attn = (const float*)d_in[4];
    const float* ln2w   = (const float*)d_in[5];
    const float* ln2b   = (const float*)d_in[6];
    const float* w_fc   = (const float*)d_in[7];
    const float* b_fc   = (const float*)d_in[8];
    const float* w_proj = (const float*)d_in[9];
    const float* b_proj = (const float*)d_in[10];
    float* out = (float*)d_out;

    char* ws = (char*)d_ws;
    size_t off = 0;
    auto alloc = [&](size_t bytes) -> void* {
        void* p = ws + off;
        off = (off + bytes + 255) & ~(size_t)255;
        return p;
    };

    float*  xln_f   = (float*) alloc((size_t)S_LEN * DMODEL * 4);
    __bf16* xln_b   = (__bf16*)alloc((size_t)S_LEN * DMODEL * 2);
    __bf16* watt_b  = (__bf16*)alloc((size_t)TD * DMODEL * 2);
    __bf16* qkv_b   = (__bf16*)alloc((size_t)S_LEN * TD * 2);
    __bf16* vT_b    = (__bf16*)alloc((size_t)DMODEL * S_LEN * 2);
    float*  ctx_f   = (float*) alloc((size_t)S_LEN * DMODEL * 4);
    float*  x2ln_f  = (float*) alloc((size_t)S_LEN * DMODEL * 4);
    __bf16* x2ln_b  = (__bf16*)alloc((size_t)S_LEN * DMODEL * 2);
    __bf16* wfc_b   = (__bf16*)alloc((size_t)DFF * DMODEL * 2);
    __bf16* hbuf_b  = (__bf16*)alloc((size_t)S_LEN * DFF * 2);
    __bf16* wproj_b = (__bf16*)alloc((size_t)DMODEL * DFF * 2);

    // 1) weights -> bf16 (stay hot in 192MB L2 across the GEMMs)
    cvt_f32_bf16<<<1024, 256, 0, stream>>>(w_attn, watt_b,  TD * DMODEL);
    cvt_f32_bf16<<<1024, 256, 0, stream>>>(w_fc,   wfc_b,   DFF * DMODEL);
    cvt_f32_bf16<<<1024, 256, 0, stream>>>(w_proj, wproj_b, DMODEL * DFF);

    // 2) LN1
    ln_fused<<<S_LEN, 256, 0, stream>>>(x, nullptr, ln1w, ln1b, xln_f, xln_b);

    // 3) QKV = ln1(x) @ w_attn^T + b_attn   [S, 3D] (bf16 for attention)
    gemm_bf16_wmma<<<dim3(S_LEN / 128, TD / 64), 256, 0, stream>>>(
        xln_b, watt_b, b_attn, nullptr, nullptr, qkv_b, S_LEN, TD, DMODEL, 0);

    // 4) V -> vT [D, S] for contiguous P.V B-fragments
    transpose_v<<<dim3(S_LEN / 64, DMODEL / 64), 256, 0, stream>>>(qkv_b, vT_b);

    // 5) attention -> ctx [S, D] fp32
    attn_flash_wmma<<<dim3(S_LEN / 64, NHEADS), 128, 0, stream>>>(qkv_b, vT_b, ctx_f);

    // 6) x2 = ln1(x) + ctx; LN2(x2)
    ln_fused<<<S_LEN, 256, 0, stream>>>(xln_f, ctx_f, ln2w, ln2b, x2ln_f, x2ln_b);

    // 7) h = gelu(x2ln @ w_fc^T + b_fc)  [S, 4D] bf16
    gemm_bf16_wmma<<<dim3(S_LEN / 128, DFF / 64), 256, 0, stream>>>(
        x2ln_b, wfc_b, b_fc, nullptr, nullptr, hbuf_b, S_LEN, DFF, DMODEL, 1);

    // 8) out = x2ln + h @ w_proj^T + b_proj  [S, D] fp32 -> d_out
    gemm_bf16_wmma<<<dim3(S_LEN / 128, DMODEL / 64), 256, 0, stream>>>(
        hbuf_b, wproj_b, b_proj, x2ln_f, out, nullptr, S_LEN, DMODEL, DFF, 0);
}
